// TimespanGuidedNeuralAttention_37477884625410
// MI455X (gfx1250) — matrix operand order repaired and checked by reading, not compiled, not run on hardware
//
#include <hip/hip_runtime.h>
#include <hip/hip_bf16.h>

#define BB 32
#define TT 2048
#define HH 512
#define UU 32
#define BT (BB * TT)

typedef __attribute__((ext_vector_type(16))) __bf16 v16bf;
typedef __attribute__((ext_vector_type(8)))  float  v8f;
typedef __attribute__((ext_vector_type(8)))  unsigned short us8;

union V16 {
    us8    h[2];
    __bf16 b[16];
    v16bf  v;
};

#if __has_builtin(__builtin_amdgcn_tanhf)
#define FAST_TANH(x) __builtin_amdgcn_tanhf(x)
#else
#define FAST_TANH(x) tanhf(x)
#endif

__device__ __forceinline__ void cvt4(V16& d, int o, float4 f) {
    d.b[o + 0] = (__bf16)f.x;   // native v_cvt (pairs into v_cvt_pk_bf16_f32)
    d.b[o + 1] = (__bf16)f.y;
    d.b[o + 2] = (__bf16)f.z;
    d.b[o + 3] = (__bf16)f.w;
}

// ---------------------------------------------------------------------------
// Kernel 0: convert + transpose weights to bf16 layouts friendly to WMMA
//   WxT, WtT : [U][H]  (B-fragment of GEMM1 reads two 8-K runs (8*hf, 16+8*hf) per lane)
//   KaT      : [H][U]  (B-fragment of GEMM2 reads two 8-U runs (8*hf, 16+8*hf) per lane)
// ---------------------------------------------------------------------------
__global__ void k_prep(const float* __restrict__ kx, const float* __restrict__ kt,
                       const float* __restrict__ ka,
                       __bf16* __restrict__ WxT,
                       __bf16* __restrict__ WtT,
                       __bf16* __restrict__ KaT) {
    int i = blockIdx.x * 256 + threadIdx.x;
    if (i < HH * UU) {
        int k = i / UU, u = i % UU;             // kernel_x/t are [H][U]
        WxT[(size_t)u * HH + k] = (__bf16)kx[i];
        WtT[(size_t)u * HH + k] = (__bf16)kt[i];
        int u2 = i / HH, hcol = i % HH;         // kernel_a is [U][H]
        KaT[(size_t)hcol * UU + u2] = (__bf16)ka[i];
    }
}

// ---------------------------------------------------------------------------
// Kernel 1: delta = lambd*tanh(x@Wx+bx) + (1-lambd)*tanh(t@Wt+bt)  (bf16 out)
// One wave per 16-row M tile; K=512 in 16 steps of v_wmma_f32_16x16x32_bf16.
// ---------------------------------------------------------------------------
__global__ __launch_bounds__(256) void k_gemm1(
        const float* __restrict__ x, const float* __restrict__ t,
        const float* __restrict__ bias_x, const float* __restrict__ bias_t,
        const float* __restrict__ lambd,
        const __bf16* __restrict__ WxT,
        const __bf16* __restrict__ WtT,
        __bf16* __restrict__ delta) {
    int lane = threadIdx.x & 31;
    int wid  = blockIdx.x * 8 + (threadIdx.x >> 5);
    int hf   = lane >> 4;          // lane half selects K sub-pattern
    int m    = lane & 15;          // A row / C column
    int rowbase = wid * 16;

    const float* xr = x + (size_t)(rowbase + m) * HH;
    const float* tr = t + (size_t)(rowbase + m) * HH;

    v8f gx0 = {}, gx1 = {}, bt0 = {}, bt1 = {};

    for (int kb = 0; kb < HH; kb += 32) {
        // A fragments (16-bit A 16x32 layout): K runs {8*hf..+8} and {16+8*hf..+8}
        V16 ax, at_;
        const float4* px0 = (const float4*)(xr + kb + 8 * hf);
        const float4* px1 = (const float4*)(xr + kb + 16 + 8 * hf);
        cvt4(ax, 0, px0[0]);  cvt4(ax, 4, px0[1]);
        cvt4(ax, 8, px1[0]);  cvt4(ax, 12, px1[1]);
        const float4* pt0 = (const float4*)(tr + kb + 8 * hf);
        const float4* pt1 = (const float4*)(tr + kb + 16 + 8 * hf);
        cvt4(at_, 0, pt0[0]); cvt4(at_, 4, pt0[1]);
        cvt4(at_, 8, pt1[0]); cvt4(at_, 12, pt1[1]);

        // B fragments: lane n = m (+16), K runs {8*hf..+8} and {16+8*hf..+8}
        const __bf16* wx0 = WxT + (size_t)m * HH + kb + 8 * hf;
        const __bf16* wx1 = WxT + (size_t)(m + 16) * HH + kb + 8 * hf;
        const __bf16* wt0 = WtT + (size_t)m * HH + kb + 8 * hf;
        const __bf16* wt1 = WtT + (size_t)(m + 16) * HH + kb + 8 * hf;
        V16 bx0, bx1, bw0, bw1;
        bx0.h[0] = *(const us8*)wx0;       bx0.h[1] = *(const us8*)(wx0 + 16);
        bx1.h[0] = *(const us8*)wx1;       bx1.h[1] = *(const us8*)(wx1 + 16);
        bw0.h[0] = *(const us8*)wt0;       bw0.h[1] = *(const us8*)(wt0 + 16);
        bw1.h[0] = *(const us8*)wt1;       bw1.h[1] = *(const us8*)(wt1 + 16);

        gx0 = __builtin_amdgcn_wmma_f32_16x16x32_bf16(false, ax.v,  false, bx0.v, (short)0, gx0, false, false);
        gx1 = __builtin_amdgcn_wmma_f32_16x16x32_bf16(false, ax.v,  false, bx1.v, (short)0, gx1, false, false);
        bt0 = __builtin_amdgcn_wmma_f32_16x16x32_bf16(false, at_.v, false, bw0.v, (short)0, bt0, false, false);
        bt1 = __builtin_amdgcn_wmma_f32_16x16x32_bf16(false, at_.v, false, bw1.v, (short)0, bt1, false, false);
    }

    float bx_lo = bias_x[m], bx_hi = bias_x[m + 16];
    float bt_lo = bias_t[m], bt_hi = bias_t[m + 16];

#pragma unroll
    for (int r = 0; r < 8; ++r) {
        int row  = rowbase + r + 8 * hf;        // C layout: M = r + 8*half
        float lam = lambd[row & (TT - 1)];      // tiles never cross a batch
        float g0 = FAST_TANH(gx0[r] + bx_lo);
        float g1 = FAST_TANH(gx1[r] + bx_hi);
        float b0 = FAST_TANH(bt0[r] + bt_lo);
        float b1 = FAST_TANH(bt1[r] + bt_hi);
        delta[(size_t)row * UU + m]      = (__bf16)(lam * g0 + (1.f - lam) * b0);
        delta[(size_t)row * UU + m + 16] = (__bf16)(lam * g1 + (1.f - lam) * b1);
    }
}

// ---------------------------------------------------------------------------
// Kernel 2: scores = delta @ kernel_a, softmax over T, fused (no scores in HBM)
// Block = (h-tile of 16, batch); 4 waves each own 512 rows of T.
// ---------------------------------------------------------------------------
__global__ __launch_bounds__(128) void k_softmax(
        const __bf16* __restrict__ delta,
        const __bf16* __restrict__ KaT,
        float* __restrict__ out) {
    __shared__ float sm[4][16];
    __shared__ float ss[4][16];

    int lane = threadIdx.x & 31;
    int wave = threadIdx.x >> 5;
    int hf   = lane >> 4;
    int nl   = lane & 15;
    int nt   = blockIdx.x;         // h tile
    int b    = blockIdx.y;         // batch

    // Constant B fragment: column (nt*16+nl) of kernel_a, U runs {8*hf..+8} and {16+8*hf..+8}
    const __bf16* kap = KaT + (size_t)(nt * 16 + nl) * UU + 8 * hf;
    V16 bfrag;
    bfrag.h[0] = *(const us8*)kap;
    bfrag.h[1] = *(const us8*)(kap + 16);

    int chunkbase = b * TT + wave * 512;
    float mmax = -3.4e38f, ssum = 0.f;

    // Pass 1: online max/sum over this wave's 32 tiles
    for (int tile = 0; tile < 32; ++tile) {
        int rowb = chunkbase + tile * 16;
        const __bf16* dr = delta + (size_t)(rowb + nl) * UU;
        V16 af;
        af.h[0] = *(const us8*)(dr + 8 * hf);
        af.h[1] = *(const us8*)(dr + 16 + 8 * hf);
        v8f z = {};
        v8f s = __builtin_amdgcn_wmma_f32_16x16x32_bf16(false, af.v, false, bfrag.v, (short)0, z, false, false);
        float tm = s[0];
#pragma unroll
        for (int r = 1; r < 8; ++r) tm = fmaxf(tm, s[r]);
        float nm  = fmaxf(mmax, tm);
        float acc = ssum * __expf(mmax - nm);
#pragma unroll
        for (int r = 0; r < 8; ++r) acc += __expf(s[r] - nm);
        mmax = nm;
        ssum = acc;
    }

    // combine lane pairs (same n column, disjoint row halves)
    float m2 = __shfl_xor(mmax, 16, 32);
    float s2 = __shfl_xor(ssum, 16, 32);
    float Mw = fmaxf(mmax, m2);
    float Sw = ssum * __expf(mmax - Mw) + s2 * __expf(m2 - Mw);
    if (lane < 16) { sm[wave][nl] = Mw; ss[wave][nl] = Sw; }
    __syncthreads();

    float Mall = -3.4e38f;
#pragma unroll
    for (int w = 0; w < 4; ++w) Mall = fmaxf(Mall, sm[w][nl]);
    float Sall = 0.f;
#pragma unroll
    for (int w = 0; w < 4; ++w) Sall += ss[w][nl] * __expf(sm[w][nl] - Mall);
    float inv = __builtin_amdgcn_rcpf(Sall);

    // Pass 2: recompute scores from L2-hot delta, emit alpha
    for (int tile = 0; tile < 32; ++tile) {
        int rowb = chunkbase + tile * 16;
        const __bf16* dr = delta + (size_t)(rowb + nl) * UU;
        V16 af;
        af.h[0] = *(const us8*)(dr + 8 * hf);
        af.h[1] = *(const us8*)(dr + 16 + 8 * hf);
        v8f z = {};
        v8f s = __builtin_amdgcn_wmma_f32_16x16x32_bf16(false, af.v, false, bfrag.v, (short)0, z, false, false);
#pragma unroll
        for (int r = 0; r < 8; ++r) {
            int row = rowb + r + 8 * hf;       // C layout row
            out[(size_t)row * HH + nt * 16 + nl] = __expf(s[r] - Mall) * inv;
        }
    }
}

extern "C" void kernel_launch(void* const* d_in, const int* in_sizes, int n_in,
                              void* d_out, int out_size, void* d_ws, size_t ws_size,
                              hipStream_t stream) {
    const float* x   = (const float*)d_in[0];
    const float* t   = (const float*)d_in[1];
    const float* kx  = (const float*)d_in[2];
    const float* kt  = (const float*)d_in[3];
    const float* ka  = (const float*)d_in[4];
    const float* bx  = (const float*)d_in[5];
    const float* bt  = (const float*)d_in[6];
    const float* lam = (const float*)d_in[7];

    __bf16* delta = (__bf16*)d_ws;                 // BT*U bf16 = 4 MB
    __bf16* WxT   = delta + (size_t)BT * UU;
    __bf16* WtT   = WxT + HH * UU;
    __bf16* KaT   = WtT + HH * UU;

    k_prep<<<(HH * UU + 255) / 256, 256, 0, stream>>>(kx, kt, ka, WxT, WtT, KaT);
    k_gemm1<<<BT / 16 / 8, 256, 0, stream>>>(x, t, bx, bt, lam, WxT, WtT, delta);
    dim3 grid(HH / 16, BB);
    k_softmax<<<grid, 128, 0, stream>>>(delta, KaT, (float*)d_out);
}
